// ChannelTransposeAttention_10307921511017
// MI455X (gfx1250) — compile-verified, matrix operand order checked by reading, not run on hardware
//
#include <hip/hip_runtime.h>
#include <hip/hip_bf16.h>

#define B_   4
#define CH   64
#define HS   64
#define WSZ  64
#define NPOS (HS*WSZ)   // 4096
#define LDP  72         // LDS row stride in halves (144B rows: 16B-aligned)

typedef __attribute__((ext_vector_type(16))) _Float16 v16h;
typedef __attribute__((ext_vector_type(8)))  float    v8f;

union Frag16 { v16h v; uint4 q[2]; };
union Pack8  { uint4 u; _Float16 h[8]; };
union Pack64 { uint4 u[8]; _Float16 h[64]; };

// Load one 16x32 f16 fragment in the A-matrix VGPR layout from a row-major
// tile.  `row` must point at (tile_row_base + this lane's row * ld).
// Lane L (r=L&15, half=L>>4) holds halves k = half*8 + {0..7} and 16+half*8+{0..7}.
__device__ __forceinline__ void load_frag(Frag16& f, const _Float16* row, int half) {
  f.q[0] = *(const uint4*)(row + half * 8);
  f.q[1] = *(const uint4*)(row + half * 8 + 16);
}

__device__ __forceinline__ v8f wmma16(const Frag16& a, const Frag16& b, v8f c) {
  return __builtin_amdgcn_wmma_f32_16x16x32_f16(false, a.v, false, b.v,
                                                (short)0, c, false, false);
}

// Async global->LDS copy, 16B per lane, tracked by ASYNCcnt (CDNA5).
// GV mode: per-lane 64-bit global address, per-lane LDS byte offset.
__device__ __forceinline__ void async_copy_b128(const void* gptr, void* lptr) {
  unsigned lds_off = (unsigned)(uintptr_t)lptr;          // addr[31:0] = LDS offset
  unsigned long long gaddr = (unsigned long long)(uintptr_t)gptr;
  asm volatile("global_load_async_to_lds_b128 %0, %1, off"
               :: "v"(lds_off), "v"(gaddr) : "memory");
}

__device__ __forceinline__ void wait_async0() {
  asm volatile("s_wait_asynccnt 0" ::: "memory");
}

// ---------------------------------------------------------------------------
// Kernel 1: per-pixel channel LayerNorm; emit xn in channel-major (for the
// depthwise convs) and position-major (WMMA B-operand for the q projection).
// ---------------------------------------------------------------------------
__global__ __launch_bounds__(256) void ln_kernel(const float* __restrict__ x,
                                                 const float* __restrict__ g,
                                                 const float* __restrict__ be,
                                                 _Float16* __restrict__ xn_c,
                                                 _Float16* __restrict__ xn_t) {
  const int idx = blockIdx.x * 256 + threadIdx.x;       // b*NPOS + p
  const int b = idx / NPOS;
  const int p = idx - b * NPOS;
  const size_t base = (size_t)b * CH * NPOS + p;
  float s = 0.f, ss = 0.f;
  for (int c = 0; c < CH; ++c) {
    float v = x[base + (size_t)c * NPOS];
    s += v; ss += v * v;
  }
  const float mu = s * (1.0f / CH);
  const float var = ss * (1.0f / CH) - mu * mu;
  const float rstd = rsqrtf(var + 1e-5f);
  Pack64 pk;
  for (int c = 0; c < CH; ++c) {
    float xv = (x[base + (size_t)c * NPOS] - mu) * rstd * g[c] + be[c];
    _Float16 hf = (_Float16)xv;
    xn_c[base + (size_t)c * NPOS] = hf;
    pk.h[c] = hf;
  }
  uint4* trow = (uint4*)(xn_t + (size_t)idx * CH);
  for (int k = 0; k < 8; ++k) trow[k] = pk.u[k];
}

// ---------------------------------------------------------------------------
// Kernel 2: depthwise 3x3 convs (k, v).  k stored position-major [p][c]
// (WMMA B operand for QK^T), v stored channel-major [c][p] (B operand for PV).
// ---------------------------------------------------------------------------
__global__ __launch_bounds__(256) void dwconv_kernel(const _Float16* __restrict__ xn_c,
                                                     const float* __restrict__ wk,
                                                     const float* __restrict__ bk,
                                                     const float* __restrict__ wv,
                                                     const float* __restrict__ bv,
                                                     _Float16* __restrict__ kT,
                                                     _Float16* __restrict__ v_c) {
  const int idx = blockIdx.x * 256 + threadIdx.x;       // over B*C*N
  const int p  = idx & (NPOS - 1);
  const int bc = idx >> 12;
  const int c  = bc & (CH - 1);
  const int b  = bc >> 6;
  const int h  = p >> 6, w = p & (WSZ - 1);
  float accK = bk[c], accV = bv[c];
  const _Float16* src = xn_c + (size_t)bc * NPOS;
  for (int dy = -1; dy <= 1; ++dy) {
    int hh = h + dy;
    if ((unsigned)hh >= HS) continue;
    for (int dx = -1; dx <= 1; ++dx) {
      int ww = w + dx;
      if ((unsigned)ww >= WSZ) continue;
      float xv = (float)src[hh * WSZ + ww];
      int wi = c * 9 + (dy + 1) * 3 + (dx + 1);
      accK += wk[wi] * xv;
      accV += wv[wi] * xv;
    }
  }
  kT[((size_t)b * NPOS + p) * CH + c] = (_Float16)accK;
  v_c[(size_t)bc * NPOS + p]          = (_Float16)accV;
}

// ---------------------------------------------------------------------------
// Kernel 3: q = wq @ xn + bq, scaled by 1/sqrt(C); stored position-major.
// 128 threads = 4 waves; each wave owns a 16-row slab of output channels.
// ---------------------------------------------------------------------------
__global__ __launch_bounds__(128) void proj_q_kernel(const float* __restrict__ wq,
                                                     const float* __restrict__ bq,
                                                     const _Float16* __restrict__ xnT,
                                                     _Float16* __restrict__ qT) {
  const int b = blockIdx.y;
  const int p0 = blockIdx.x * 64;
  const int wave = threadIdx.x >> 5;
  const int lane = threadIdx.x & 31;
  const int half = lane >> 4, r = lane & 15;
  const int co0 = wave * 16;

  Frag16 a[2];
  const float* wrow = wq + (co0 + r) * CH;
  for (int ck = 0; ck < 2; ++ck)
    for (int j = 0; j < 8; ++j) {
      a[ck].v[j]     = (_Float16)wrow[ck * 32 + half * 8 + j];
      a[ck].v[j + 8] = (_Float16)wrow[ck * 32 + half * 8 + 16 + j];
    }
  float bias[8];
  for (int v = 0; v < 8; ++v) bias[v] = bq[co0 + 8 * half + v];

  for (int pt = 0; pt < 4; ++pt) {
    const int p1 = p0 + pt * 16;
    const _Float16* xrow = xnT + ((size_t)b * NPOS + p1 + r) * CH;
    Frag16 bm0, bm1;
    load_frag(bm0, xrow, half);
    load_frag(bm1, xrow + 32, half);
    v8f acc = {};
    acc = wmma16(a[0], bm0, acc);
    acc = wmma16(a[1], bm1, acc);
    Pack8 pk;
    for (int v = 0; v < 8; ++v)
      pk.h[v] = (_Float16)((acc[v] + bias[v]) * 0.125f);   // fold 1/sqrt(64)
    *(uint4*)(qT + ((size_t)b * NPOS + p1 + r) * CH + co0 + 8 * half) = pk.u;
  }
}

// ---------------------------------------------------------------------------
// Kernel 4: fused flash attention.  256 threads = 8 waves, each wave owns a
// 16-query tile (128 queries/block).  Key/value blocks of 64 staged in LDS by
// ASYNCcnt-tracked global_load_async_to_lds_b128, double-buffered so the next
// tile's DMA overlaps the current tile's 16 WMMAs.
// ---------------------------------------------------------------------------
__global__ __launch_bounds__(256) void attn_kernel(const _Float16* __restrict__ qT,
                                                   const _Float16* __restrict__ kT,
                                                   const _Float16* __restrict__ v_c,
                                                   _Float16* __restrict__ outT) {
  __shared__ _Float16 lds_k[2][64 * LDP];
  __shared__ _Float16 lds_v[2][64 * LDP];
  __shared__ _Float16 lds_p[8 * 16 * LDP];

  const int b = blockIdx.y;
  const int qbase = blockIdx.x * 128;
  const int tid = threadIdx.x;
  const int wave = tid >> 5;
  const int lane = tid & 31;
  const int half = lane >> 4, r = lane & 15;
  const int i0 = qbase + wave * 16;

  // Q fragments (A operand), fixed for the whole block loop.
  const _Float16* qrow = qT + ((size_t)b * NPOS + i0 + r) * CH;
  Frag16 qa0, qa1;
  load_frag(qa0, qrow, half);
  load_frag(qa1, qrow + 32, half);

  float mi[8], li[8];
  v8f o[4];
  for (int v = 0; v < 8; ++v) { mi[v] = -3.0e38f; li[v] = 0.f; }
  for (int t = 0; t < 4; ++t) o[t] = (v8f){};

  _Float16* prow_base = lds_p + wave * 16 * LDP;

  // Issue async staging of tile `kb` into buffer `buf` (4 async ops / wave).
  auto issue_tile = [&](int kb, int buf) {
    const int j0 = kb * 64;
    const _Float16* kbase = kT + ((size_t)b * NPOS + j0) * CH;
    for (int it = 0; it < 2; ++it) {
      int idx = tid + it * 256;                 // 0..511 b128 chunks
      int row = idx >> 3, col = idx & 7;
      async_copy_b128(kbase + row * CH + col * 8,
                      lds_k[buf] + row * LDP + col * 8);
    }
    for (int it = 0; it < 2; ++it) {
      int idx = tid + it * 256;
      int row = idx >> 3, col = idx & 7;
      async_copy_b128(v_c + ((size_t)b * CH + row) * NPOS + j0 + col * 8,
                      lds_v[buf] + row * LDP + col * 8);
    }
  };

  issue_tile(0, 0);

  for (int kb = 0; kb < NPOS / 64; ++kb) {
    const int cur = kb & 1;
    wait_async0();        // my 4 copies for buffer `cur` have landed
    __syncthreads();      // everyone's copies landed; everyone done reading `cur^1`
    if (kb + 1 < NPOS / 64) issue_tile(kb + 1, cur ^ 1);

    // S = Q K^T : 4 key sub-tiles of 16, contraction over 64 channels.
    v8f s[4];
    for (int t = 0; t < 4; ++t) {
      const _Float16* krow = lds_k[cur] + (t * 16 + r) * LDP;
      Frag16 kb0, kb1;
      load_frag(kb0, krow, half);
      load_frag(kb1, krow + 32, half);
      v8f acc = {};
      acc = wmma16(qa0, kb0, acc);
      acc = wmma16(qa1, kb1, acc);
      s[t] = acc;
    }

    // Online softmax; D-layout row m = v + 8*half lives across 16 lanes.
    for (int v = 0; v < 8; ++v) {
      float rm = fmaxf(fmaxf(s[0][v], s[1][v]), fmaxf(s[2][v], s[3][v]));
      for (int msk = 1; msk < 16; msk <<= 1)
        rm = fmaxf(rm, __shfl_xor(rm, msk, 32));
      float mnew = fmaxf(mi[v], rm);
      float corr = __expf(mi[v] - mnew);
      float rs = 0.f;
      for (int t = 0; t < 4; ++t) {
        float pv = __expf(s[t][v] - mnew);
        s[t][v] = pv;
        rs += pv;
      }
      for (int msk = 1; msk < 16; msk <<= 1)
        rs += __shfl_xor(rs, msk, 32);
      li[v] = li[v] * corr + rs;
      mi[v] = mnew;
      for (int t = 0; t < 4; ++t) o[t][v] *= corr;
      // Scatter P row (m = v + 8*half) into this wave's LDS region.
      const int rowm = v + 8 * half;
      for (int t = 0; t < 4; ++t)
        prow_base[rowm * LDP + t * 16 + r] = (_Float16)s[t][v];
    }
    // Same-wave LDS RAW: drain DS counter before re-reading P as A fragments.
    asm volatile("s_wait_dscnt 0" ::: "memory");

    // O += P V : A = P (16 x 64 probs), B = V channel-tiles.
    Frag16 pa0, pa1;
    const _Float16* prow = prow_base + r * LDP;
    load_frag(pa0, prow, half);
    load_frag(pa1, prow + 32, half);
    for (int t = 0; t < 4; ++t) {
      const _Float16* vrow = lds_v[cur] + (t * 16 + r) * LDP;
      Frag16 vb0, vb1;
      load_frag(vb0, vrow, half);
      load_frag(vb1, vrow + 32, half);
      o[t] = wmma16(pa0, vb0, o[t]);
      o[t] = wmma16(pa1, vb1, o[t]);
    }
  }

  // Normalize and store O position-major [i][c] as f16.
  for (int v = 0; v < 8; ++v) {
    const float inv = 1.0f / li[v];
    const int rowi = i0 + v + 8 * half;
    for (int t = 0; t < 4; ++t)
      outT[((size_t)b * NPOS + rowi) * CH + t * 16 + r] = (_Float16)(o[t][v] * inv);
  }
}

// ---------------------------------------------------------------------------
// Kernel 5: out = wo @ O + bo + x   (f32 output, channel-major)
// ---------------------------------------------------------------------------
__global__ __launch_bounds__(128) void proj_o_kernel(const float* __restrict__ wo,
                                                     const float* __restrict__ bo,
                                                     const _Float16* __restrict__ outT,
                                                     const float* __restrict__ x,
                                                     float* __restrict__ out) {
  const int b = blockIdx.y;
  const int p0 = blockIdx.x * 64;
  const int wave = threadIdx.x >> 5;
  const int lane = threadIdx.x & 31;
  const int half = lane >> 4, r = lane & 15;
  const int co0 = wave * 16;

  Frag16 a[2];
  const float* wrow = wo + (co0 + r) * CH;
  for (int ck = 0; ck < 2; ++ck)
    for (int j = 0; j < 8; ++j) {
      a[ck].v[j]     = (_Float16)wrow[ck * 32 + half * 8 + j];
      a[ck].v[j + 8] = (_Float16)wrow[ck * 32 + half * 8 + 16 + j];
    }
  float bias[8];
  for (int v = 0; v < 8; ++v) bias[v] = bo[co0 + 8 * half + v];

  for (int pt = 0; pt < 4; ++pt) {
    const int p1 = p0 + pt * 16;
    const _Float16* orow = outT + ((size_t)b * NPOS + p1 + r) * CH;
    Frag16 bm0, bm1;
    load_frag(bm0, orow, half);
    load_frag(bm1, orow + 32, half);
    v8f acc = {};
    acc = wmma16(a[0], bm0, acc);
    acc = wmma16(a[1], bm1, acc);
    for (int v = 0; v < 8; ++v) {
      const int co = co0 + v + 8 * half;
      const size_t oidx = ((size_t)b * CH + co) * NPOS + p1 + r;
      out[oidx] = acc[v] + bias[v] + x[oidx];
    }
  }
}

// ---------------------------------------------------------------------------
extern "C" void kernel_launch(void* const* d_in, const int* in_sizes, int n_in,
                              void* d_out, int out_size, void* d_ws, size_t ws_size,
                              hipStream_t stream) {
  const float* x   = (const float*)d_in[0];
  const float* lng = (const float*)d_in[1];
  const float* lnb = (const float*)d_in[2];
  const float* wq  = (const float*)d_in[3];
  const float* bq  = (const float*)d_in[4];
  const float* wk  = (const float*)d_in[5];
  const float* bk  = (const float*)d_in[6];
  const float* wv  = (const float*)d_in[7];
  const float* bv  = (const float*)d_in[8];
  const float* wo  = (const float*)d_in[9];
  const float* bo  = (const float*)d_in[10];
  float* out = (float*)d_out;

  char* ws = (char*)d_ws;
  const size_t T = (size_t)B_ * CH * NPOS * sizeof(_Float16);   // 2 MB each
  _Float16* xn_c = (_Float16*)(ws + 0 * T);
  _Float16* xn_t = (_Float16*)(ws + 1 * T);
  _Float16* qT   = (_Float16*)(ws + 2 * T);
  _Float16* kT   = (_Float16*)(ws + 3 * T);
  _Float16* v_c  = (_Float16*)(ws + 4 * T);
  _Float16* outT = (_Float16*)(ws + 5 * T);

  ln_kernel<<<dim3((B_ * NPOS) / 256), 256, 0, stream>>>(x, lng, lnb, xn_c, xn_t);
  dwconv_kernel<<<dim3((B_ * CH * NPOS) / 256), 256, 0, stream>>>(xn_c, wk, bk, wv, bv, kT, v_c);
  proj_q_kernel<<<dim3(NPOS / 64, B_), 128, 0, stream>>>(wq, bq, xn_t, qT);
  attn_kernel<<<dim3(NPOS / 128, B_), 256, 0, stream>>>(qT, kT, v_c, outT);
  proj_o_kernel<<<dim3(NPOS / 64, B_), 128, 0, stream>>>(wo, bo, outT, x, out);
}
